// SplitBrainAttention_20461224198454
// MI455X (gfx1250) — matrix operand order repaired and checked by reading, not compiled, not run on hardware
//
#include <hip/hip_runtime.h>
#include <hip/hip_bf16.h>
#include <math.h>

typedef __bf16 bf16;
typedef __attribute__((ext_vector_type(16))) __bf16 v16bf;
typedef __attribute__((ext_vector_type(8)))  __bf16 v8bf;
typedef __attribute__((ext_vector_type(8)))  float  v8f;

#define WMMA_BF16(a, b, c) \
  __builtin_amdgcn_wmma_f32_16x16x32_bf16(false, (a), false, (b), (short)0, (c), false, false)

// Problem constants (match reference)
constexpr int Bb  = 2;
constexpr int Ll  = 2048;
constexpr int Dd  = 2048;
constexpr int Nn  = 8;
constexpr int KVh = 4;
constexpr int Hh  = 256;
constexpr float SOFTCAP = 50.0f;
constexpr float KMASK   = -2.3819763e38f;

// Concatenate two 16-byte bf16 chunks into a WMMA fragment register set.
static __device__ inline v16bf frag16(const bf16* p0, const bf16* p1) {
  v8bf a = *(const v8bf*)p0;
  v8bf b = *(const v8bf*)p1;
  return __builtin_shufflevector(a, b, 0, 1, 2, 3, 4, 5, 6, 7,
                                 8, 9, 10, 11, 12, 13, 14, 15);
}

// Native gfx1250 transcendental tanh (V_TANH_F32). Trailing v_nops cover the
// documented TRANS-op result-use hazard since the compiler can't see into asm.
static __device__ inline float tanh_fast(float x) {
  float r;
  asm volatile("v_tanh_f32 %0, %1\n\tv_nop\n\tv_nop" : "=v"(r) : "v"(x));
  return r;
}

// ---------------------------------------------------------------------------
// Tiled GEMM: C[M x Nc] = A[M x K] * B[K x Nc], f32 in/out, bf16 WMMA compute.
// Block = 256 threads (8 waves), 128x64 C tile, K stepped by 32 through LDS.
// Each wave owns a 32x32 sub-tile: 2 A-frags x 2 B-frags -> 4 WMMAs per
// k-step between barriers. A staged row-major (k-contiguous); B staged
// TRANSPOSED so every fragment is two ds_load_b128 per lane. Next k-slab is
// prefetched (global_prefetch_b8) while the current one is consumed.
// ---------------------------------------------------------------------------
__global__ __launch_bounds__(256) void gemm_f32_bf16(const float* __restrict__ A,
                                                     const float* __restrict__ Bm,
                                                     float* __restrict__ C,
                                                     int M, int Ncols, int K) {
  __shared__ __align__(16) bf16 As[128][40];  // row stride 80B: 16B-aligned, conflict-free
  __shared__ __align__(16) bf16 Bst[64][40];  // [n][k] transposed

  const int tid  = threadIdx.x;
  const int wid  = tid >> 5;
  const int lane = tid & 31;
  const int lm   = lane & 15;
  const int lh   = lane >> 4;
  const int m0   = blockIdx.y * 128;
  const int n0   = blockIdx.x * 64;
  const int tr   = wid & 3;          // wave's 32-row strip of the 128-row tile
  const int cs   = (wid >> 2) * 32;  // wave's 32-col strip

  // fixed staging assignments (no guarded loops)
  const int am = tid >> 1, ak = (tid & 1) * 16;  // A: thread -> (row, 16 k's)
  const int bk = tid >> 3, bn = (tid & 7) * 8;   // B: thread -> (k, 8 n's)

  v8f zero = {0.f, 0.f, 0.f, 0.f, 0.f, 0.f, 0.f, 0.f};
  v8f c00 = zero, c01 = zero, c10 = zero, c11 = zero;

  for (int k0 = 0; k0 < K; k0 += 32) {
    // Stage A(128x32): 4x float4 loads -> two 16B LDS stores per thread.
    {
      const float4* src = (const float4*)&A[(size_t)(m0 + am) * K + k0 + ak];
      float4 a0 = src[0], a1 = src[1], a2 = src[2], a3 = src[3];
      v8bf s0, s1;
      s0[0] = (bf16)a0.x; s0[1] = (bf16)a0.y; s0[2] = (bf16)a0.z; s0[3] = (bf16)a0.w;
      s0[4] = (bf16)a1.x; s0[5] = (bf16)a1.y; s0[6] = (bf16)a1.z; s0[7] = (bf16)a1.w;
      s1[0] = (bf16)a2.x; s1[1] = (bf16)a2.y; s1[2] = (bf16)a2.z; s1[3] = (bf16)a2.w;
      s1[4] = (bf16)a3.x; s1[5] = (bf16)a3.y; s1[6] = (bf16)a3.z; s1[7] = (bf16)a3.w;
      *(v8bf*)&As[am][ak]     = s0;
      *(v8bf*)&As[am][ak + 8] = s1;
    }
    // Stage B(32x64) transposed: 2x float4 loads -> 8 scattered b16 stores.
    {
      const float4* src = (const float4*)&Bm[(size_t)(k0 + bk) * Ncols + n0 + bn];
      float4 b0 = src[0], b1 = src[1];
      Bst[bn + 0][bk] = (bf16)b0.x; Bst[bn + 1][bk] = (bf16)b0.y;
      Bst[bn + 2][bk] = (bf16)b0.z; Bst[bn + 3][bk] = (bf16)b0.w;
      Bst[bn + 4][bk] = (bf16)b1.x; Bst[bn + 5][bk] = (bf16)b1.y;
      Bst[bn + 6][bk] = (bf16)b1.z; Bst[bn + 7][bk] = (bf16)b1.w;
    }
    // Prefetch next k-slab while this one is consumed (speculative, safe).
    {
      int kp = (k0 + 32 < K) ? (k0 + 32) : k0;
      __builtin_prefetch(&A[(size_t)(m0 + am) * K + kp + ak], 0, 3);
      __builtin_prefetch(&Bm[(size_t)(kp + bk) * Ncols + n0 + bn], 0, 3);
    }
    __syncthreads();

    // Fragments: A element e -> k = lh*16 + e          (two 16B runs);
    //            B element e -> k = lh*16 + e          (two 16B runs).
    const int ra = tr * 32;
    v16bf af0 = frag16(&As[ra + lm][lh * 16],      &As[ra + lm][lh * 16 + 8]);
    v16bf af1 = frag16(&As[ra + 16 + lm][lh * 16], &As[ra + 16 + lm][lh * 16 + 8]);
    v16bf bf0 = frag16(&Bst[cs + lm][lh * 16],      &Bst[cs + lm][lh * 16 + 8]);
    v16bf bf1 = frag16(&Bst[cs + 16 + lm][lh * 16], &Bst[cs + 16 + lm][lh * 16 + 8]);
    c00 = WMMA_BF16(af0, bf0, c00);
    c01 = WMMA_BF16(af0, bf1, c01);
    c10 = WMMA_BF16(af1, bf0, c10);
    c11 = WMMA_BF16(af1, bf1, c11);
    __syncthreads();
  }

  #pragma unroll
  for (int r = 0; r < 8; ++r) {
    int m = m0 + tr * 32 + r + 8 * lh;               // C layout: m = r + 8*(lane/16)
    C[(size_t)m * Ncols + n0 + cs + lm]             = c00[r];
    C[(size_t)m * Ncols + n0 + cs + 16 + lm]        = c01[r];
    C[(size_t)(m + 16) * Ncols + n0 + cs + lm]      = c10[r];
    C[(size_t)(m + 16) * Ncols + n0 + cs + 16 + lm] = c11[r];
  }
}

// ---------------------------------------------------------------------------
// RoPE + optional query scaling + f32 -> bf16. One block per (b, t, head).
// ---------------------------------------------------------------------------
__global__ __launch_bounds__(128) void rope_cvt(const float* __restrict__ raw,
                                                bf16* __restrict__ outb,
                                                const int* __restrict__ segpos,
                                                int nheads, float scale) {
  const int idx = blockIdx.x;
  const int n   = idx % nheads;
  const int bt  = idx / nheads;         // b*L + t
  const int i   = threadIdx.x;          // 0..127 (half of H)

  float fraction = (2.0f * (float)i) / 256.0f;
  float ts  = powf(10000.0f, fraction);
  float pos = (float)segpos[bt];
  float ang = pos / ts;
  float sn  = __sinf(ang), cs = __cosf(ang);

  size_t base = ((size_t)bt * nheads + n) * 256;
  float x1 = raw[base + i];
  float x2 = raw[base + 128 + i];
  outb[base + i]       = (bf16)((x1 * cs - x2 * sn) * scale);
  outb[base + 128 + i] = (bf16)((x2 * cs + x1 * sn) * scale);
}

// ---------------------------------------------------------------------------
// V transpose: [B*L, KV, H] f32 -> [B, KV, H, L] bf16 (sequence-contiguous)
// so P*V B-fragments are contiguous 16-byte loads. 32x32 tiles via LDS.
// Grid: (L/32, H/32, B*KV), 256 threads.
// ---------------------------------------------------------------------------
__global__ __launch_bounds__(256) void transpose_v(const float* __restrict__ in,
                                                   bf16* __restrict__ out) {
  __shared__ float tile[32][33];
  const int bkv = blockIdx.z;
  const int b = bkv / KVh, kv = bkv % KVh;
  const int t0 = blockIdx.x * 32;
  const int h0 = blockIdx.y * 32;
  const int tx = threadIdx.x & 31, ty = threadIdx.x >> 5;  // ty = 0..7
  #pragma unroll
  for (int j = 0; j < 4; ++j) {
    int t = t0 + ty + j * 8;
    tile[ty + j * 8][tx] = in[((size_t)(b * Ll + t) * KVh + kv) * Hh + h0 + tx];
  }
  __syncthreads();
  #pragma unroll
  for (int j = 0; j < 4; ++j) {
    int h = h0 + ty + j * 8;
    out[((size_t)bkv * Hh + h) * Ll + t0 + tx] = (bf16)tile[tx][ty + j * 8];
  }
}

// ---------------------------------------------------------------------------
// Flash attention, soft-cap tanh + causal (+ student random) mask.
// Grid: (L/64, N, B). Block = 128 (4 waves); each wave owns 16 query rows,
// full H=256 accumulator in registers. 32 keys/step: 16 WMMA (QK^T) +
// online softmax (lane shuffles, native v_tanh) + 16 WMMA (P*V).
// ---------------------------------------------------------------------------
__global__ __launch_bounds__(128) void flash_attn(const bf16* __restrict__ qb,
                                                  const bf16* __restrict__ kb,
                                                  const bf16* __restrict__ vbT,
                                                  float* __restrict__ ao,
                                                  const unsigned char* __restrict__ key_keep,
                                                  int student) {
  const int b    = blockIdx.z;
  const int n    = blockIdx.y;
  const int wid  = threadIdx.x >> 5;
  const int lane = threadIdx.x & 31;
  const int lm   = lane & 15;
  const int lh   = lane >> 4;
  const int q0   = blockIdx.x * 64 + wid * 16;
  const int kvh  = n >> 1;  // GQA: N/KV = 2

  __shared__ __align__(16) bf16 Plds[4][16][40];

  // Q fragments for this wave's 16 rows: two contiguous 16B runs per chunk.
  v16bf qf[8];
  #pragma unroll
  for (int c = 0; c < 8; ++c) {
    const bf16* qp = qb + ((size_t)(b * Ll + q0 + lm) * Nn + n) * Hh + c * 32 + lh * 16;
    qf[c] = frag16(qp, qp + 8);
  }

  v8f zero = {0.f, 0.f, 0.f, 0.f, 0.f, 0.f, 0.f, 0.f};
  v8f acc[16];
  #pragma unroll
  for (int hc = 0; hc < 16; ++hc) acc[hc] = zero;
  float mrun[8], lrun[8];
  #pragma unroll
  for (int r = 0; r < 8; ++r) { mrun[r] = -INFINITY; lrun[r] = 0.f; }

  const unsigned char* kk = key_keep + (size_t)b * Ll;  // key_keep[b, 0, :]

  for (int s0 = 0; s0 < q0 + 16; s0 += 32) {
    // Prefetch next step's K/V rows for this lane (speculative).
    {
      int sp = (s0 + 32 < q0 + 16) ? (s0 + 32) : s0;
      __builtin_prefetch(kb + ((size_t)(b * Ll + sp + lm) * KVh + kvh) * Hh, 0, 3);
      __builtin_prefetch(vbT + ((size_t)((b * KVh + kvh) * Hh) + lane * 8) * Ll + sp, 0, 3);
    }

    // ---- logits: two 16x16 tiles over 32 keys, K=H=256 in 8 WMMA steps ----
    v8f sA = zero, sB = zero;
    #pragma unroll
    for (int c = 0; c < 8; ++c) {
      const bf16* k0p = kb + ((size_t)(b * Ll + s0 + lm) * KVh + kvh) * Hh + c * 32 + lh * 16;
      const bf16* k1p = kb + ((size_t)(b * Ll + s0 + 16 + lm) * KVh + kvh) * Hh + c * 32 + lh * 16;
      sA = WMMA_BF16(qf[c], frag16(k0p, k0p + 8), sA);
      sB = WMMA_BF16(qf[c], frag16(k1p, k1p + 8), sB);
    }

    // ---- soft-cap, mask, online softmax ----
    #pragma unroll
    for (int r = 0; r < 8; ++r) {
      int qpos = q0 + r + 8 * lh;
      int sp0 = s0 + lm, sp1 = s0 + 16 + lm;
      float x0 = SOFTCAP * tanh_fast(sA[r] * (1.0f / SOFTCAP));
      float x1 = SOFTCAP * tanh_fast(sB[r] * (1.0f / SOFTCAP));
      bool keep0 = (sp0 <= qpos) && (!student || kk[sp0] != 0 || sp0 == qpos);
      bool keep1 = (sp1 <= qpos) && (!student || kk[sp1] != 0 || sp1 == qpos);
      x0 = keep0 ? x0 : KMASK;
      x1 = keep1 ? x1 : KMASK;

      float vmax = fmaxf(x0, x1);
      vmax = fmaxf(vmax, __shfl_xor(vmax, 1));
      vmax = fmaxf(vmax, __shfl_xor(vmax, 2));
      vmax = fmaxf(vmax, __shfl_xor(vmax, 4));
      vmax = fmaxf(vmax, __shfl_xor(vmax, 8));

      float mnew = fmaxf(mrun[r], vmax);
      float sc = __expf(mrun[r] - mnew);
      float p0 = keep0 ? __expf(x0 - mnew) : 0.f;   // explicit zero: fully-masked
      float p1 = keep1 ? __expf(x1 - mnew) : 0.f;   // tiles can't pollute sums
      float rs = p0 + p1;
      rs += __shfl_xor(rs, 1);
      rs += __shfl_xor(rs, 2);
      rs += __shfl_xor(rs, 4);
      rs += __shfl_xor(rs, 8);

      lrun[r] = lrun[r] * sc + rs;
      mrun[r] = mnew;
      #pragma unroll
      for (int hc = 0; hc < 16; ++hc) acc[hc][r] *= sc;

      // stash P (C layout) so it can be re-read in A-fragment layout
      Plds[wid][r + 8 * lh][lm]      = (bf16)p0;
      Plds[wid][r + 8 * lh][16 + lm] = (bf16)p1;
    }
    __builtin_amdgcn_wave_barrier();  // order wave-local LDS store -> load

    v16bf pf = frag16(&Plds[wid][lm][lh * 16], &Plds[wid][lm][lh * 16 + 8]);

    // ---- P * V over 32 keys into 16 h-chunks (V is seq-contiguous) ----
    #pragma unroll
    for (int hc = 0; hc < 16; ++hc) {
      const bf16* vp = vbT + ((size_t)((b * KVh + kvh) * Hh) + hc * 16 + lm) * Ll
                           + s0 + lh * 16;
      acc[hc] = WMMA_BF16(pf, frag16(vp, vp + 8), acc[hc]);
    }
  }

  // ---- finalize: divide by softmax denominator, write [B*L, N, H] f32 ----
  #pragma unroll
  for (int hc = 0; hc < 16; ++hc) {
    #pragma unroll
    for (int r = 0; r < 8; ++r) {
      size_t o = ((size_t)(b * Ll + q0 + r + 8 * lh) * Nn + n) * Hh + hc * 16 + lm;
      ao[o] = acc[hc][r] / lrun[r];
    }
  }
}

// ---------------------------------------------------------------------------
extern "C" void kernel_launch(void* const* d_in, const int* in_sizes, int n_in,
                              void* d_out, int out_size, void* d_ws, size_t ws_size,
                              hipStream_t stream) {
  (void)in_sizes; (void)n_in; (void)out_size; (void)ws_size;

  const float* x       = (const float*)d_in[0];
  const int*   segpos  = (const int*)d_in[1];
  // d_in[2] = attn_mask (pure causal tril -> computed analytically)
  const unsigned char* key_keep = (const unsigned char*)d_in[3];  // [B,1,L] bool
  const float* wq_t = (const float*)d_in[4];
  const float* wk_t = (const float*)d_in[5];
  const float* wv_t = (const float*)d_in[6];
  const float* wo_t = (const float*)d_in[7];
  const float* wq_s = (const float*)d_in[8];
  const float* wk_s = (const float*)d_in[9];
  const float* wv_s = (const float*)d_in[10];
  const float* wo_s = (const float*)d_in[11];

  constexpr size_t M  = (size_t)Bb * Ll;          // 4096 rows
  constexpr size_t QE = M * (size_t)Nn * Hh;      // q elems  = 4096*2048
  constexpr size_t KE = M * (size_t)KVh * Hh;     // kv elems = 4096*1024

  // Workspace layout: f32 staging then bf16 buffers. Attention output reuses
  // the q f32 staging region (already consumed by rope_cvt at that point).
  float* qraw_t = (float*)d_ws;
  float* kraw_t = qraw_t + QE;
  float* vraw_t = kraw_t + KE;
  float* qraw_s = vraw_t + KE;
  float* kraw_s = qraw_s + QE;
  float* vraw_s = kraw_s + KE;
  bf16*  qb_t   = (bf16*)(vraw_s + KE);
  bf16*  kb_t   = qb_t + QE;
  bf16*  vb_t   = kb_t + KE;   // transposed [B,KV,H,L]
  bf16*  qb_s   = vb_t + KE;
  bf16*  kb_s   = qb_s + QE;
  bf16*  vb_s   = kb_s + KE;   // transposed
  float* ao_t   = qraw_t;      // reuse
  float* ao_s   = qraw_s;      // reuse

  dim3 blk(256);
  dim3 gq(Dd / 64, (int)(M / 128));            // 32 x 32  (Nc = N*H = 2048)
  dim3 gk((KVh * Hh) / 64, (int)(M / 128));    // 16 x 32  (Nc = KV*H = 1024)

  // 1) QKV projections (teacher + student)
  gemm_f32_bf16<<<gq, blk, 0, stream>>>(x, wq_t, qraw_t, (int)M, Nn * Hh, Dd);
  gemm_f32_bf16<<<gk, blk, 0, stream>>>(x, wk_t, kraw_t, (int)M, KVh * Hh, Dd);
  gemm_f32_bf16<<<gk, blk, 0, stream>>>(x, wv_t, vraw_t, (int)M, KVh * Hh, Dd);
  gemm_f32_bf16<<<gq, blk, 0, stream>>>(x, wq_s, qraw_s, (int)M, Nn * Hh, Dd);
  gemm_f32_bf16<<<gk, blk, 0, stream>>>(x, wk_s, kraw_s, (int)M, KVh * Hh, Dd);
  gemm_f32_bf16<<<gk, blk, 0, stream>>>(x, wv_s, vraw_s, (int)M, KVh * Hh, Dd);

  // 2) RoPE (+ 1/16 query scale), bf16 conversion, V transpose
  rope_cvt<<<dim3((unsigned)(M * Nn)), dim3(128), 0, stream>>>(qraw_t, qb_t, segpos, Nn, 0.0625f);
  rope_cvt<<<dim3((unsigned)(M * KVh)), dim3(128), 0, stream>>>(kraw_t, kb_t, segpos, KVh, 1.0f);
  rope_cvt<<<dim3((unsigned)(M * Nn)), dim3(128), 0, stream>>>(qraw_s, qb_s, segpos, Nn, 0.0625f);
  rope_cvt<<<dim3((unsigned)(M * KVh)), dim3(128), 0, stream>>>(kraw_s, kb_s, segpos, KVh, 1.0f);
  dim3 gt(Ll / 32, Hh / 32, Bb * KVh);
  transpose_v<<<gt, dim3(256), 0, stream>>>(vraw_t, vb_t);
  transpose_v<<<gt, dim3(256), 0, stream>>>(vraw_s, vb_s);

  // 3) Flash attention (teacher: causal only; student: causal & (keep | eye))
  dim3 ga(Ll / 64, Nn, Bb);
  flash_attn<<<ga, dim3(128), 0, stream>>>(qb_t, kb_t, vb_t, ao_t, key_keep, 0);
  flash_attn<<<ga, dim3(128), 0, stream>>>(qb_s, kb_s, vb_s, ao_s, key_keep, 1);

  // 4) Output projections straight into d_out (teacher then student)
  float* outp = (float*)d_out;
  gemm_f32_bf16<<<gq, blk, 0, stream>>>(ao_t, wo_t, outp, (int)M, Dd, Nn * Hh);
  gemm_f32_bf16<<<gq, blk, 0, stream>>>(ao_s, wo_s, outp + M * Dd, (int)M, Dd, Nn * Hh);
}
